// LSTM_19679540150297
// MI455X (gfx1250) — matrix operand order repaired and checked by reading, not compiled, not run on hardware
//
#include <hip/hip_runtime.h>
#include <stdint.h>

typedef __attribute__((ext_vector_type(16))) __bf16 v16bf;
typedef __attribute__((ext_vector_type(8)))  float  v8f;

#define HDIM    256
#define TSTEPS  128
#define NCLS    10
#define TILE_B  32            // batch rows per workgroup (2 WMMA m-tiles)
#define NKT_LDS 4             // k-slices of weights cached in LDS (4/8 = 256 KB)

__device__ __forceinline__ uint32_t f32_to_bf16_bits(float f) {
    union { float f; uint32_t u; } c; c.f = f;
    uint32_t u = c.u;
    return ((u + 0x7FFFu + ((u >> 16) & 1u)) >> 16) & 0xFFFFu;   // RNE
}

// CDNA5 hardware v_tanh_f32 (TRANS op).
__device__ __forceinline__ float hw_tanh(float x) {
#if __has_builtin(__builtin_amdgcn_tanhf)
    return __builtin_amdgcn_tanhf(x);
#else
    float y;
    asm("v_tanh_f32 %0, %1\n\tv_nop" : "=v"(y) : "v"(x));
    return y;
#endif
}
__device__ __forceinline__ float fast_sig(float x) {
    return fmaf(0.5f, hw_tanh(0.5f * x), 0.5f);   // exact identity
}

// Fragment union: 2x16B raw loads <-> WMMA bf16 operand
union ABu {
    uint4 u[2];
    v16bf v;
};

// ---------------------------------------------------------------------------
// Repack the 4 recurrent weight matrices (H,H fp32, row=k, col=j) into bf16
// WMMA B-fragment order. fragment(nt,kt)=1KB: n = nt*16 + lane%16,
// lanes0-15: K=kt*32+{0..7,16..23}, lanes16-31: K=kt*32+{8..15,24..31}, even K in [15:0].
// Per-lane bytes contiguous (lane*32..+31) -> coalesced b128 loads.
// ---------------------------------------------------------------------------
__global__ __launch_bounds__(256) void repack_wh(
    const float* __restrict__ wg, const float* __restrict__ wi,
    const float* __restrict__ wf, const float* __restrict__ wo,
    uint32_t* __restrict__ frag)
{
    int gidx    = blockIdx.x * 256 + threadIdx.x;   // one dword each
    int fragIdx = gidx >> 8;                        // 256 dwords / fragment
    int within  = gidx & 255;
    int lane    = within >> 3;
    int v       = within & 7;
    int nt      = fragIdx >> 3;
    int kt      = fragIdx & 7;
    int k       = kt * 32 + ((lane < 16) ? 0 : 8) + 2 * v + ((v < 4) ? 0 : 8);
    int n       = nt * 16 + (lane & 15);
    int gate    = n >> 8;
    int j       = n & 255;
    const float* w = (gate == 0) ? wg : (gate == 1) ? wi : (gate == 2) ? wf : wo;
    float e0 = w[k * HDIM + j];
    float e1 = w[(k + 1) * HDIM + j];
    frag[gidx] = f32_to_bf16_bits(e0) | (f32_to_bf16_bits(e1) << 16);
}

// ---------------------------------------------------------------------------
// Persistent LSTM: 32 workgroups x 256 threads (8 wave32).
// Each WG owns 32 batch rows for all T steps (recurrence independent per
// batch row -> no inter-WG sync). Wave w owns hidden units j in [32w,32w+32)
// for all 4 gates -> gate math and cell state stay wave-local in VGPRs.
// Weights: 4 k-slices cached in LDS (256 KB), 4 streamed from L2 each step.
// kt=0 is peeled so its 16 WMMAs take the inline-0 C operand (no per-step
// accumulator zeroing). Opaque zero offsets defeat loop-invariant hoisting.
// ---------------------------------------------------------------------------
__global__ __launch_bounds__(256) void lstm_persist(
    const float* __restrict__ x,
    const float* __restrict__ w_gx, const float* __restrict__ w_ix,
    const float* __restrict__ w_fx, const float* __restrict__ w_ox,
    const float* __restrict__ b_g,  const float* __restrict__ b_i,
    const float* __restrict__ b_f,  const float* __restrict__ b_o,
    const float* __restrict__ w_ph, const float* __restrict__ b_p,
    const uint32_t* __restrict__ wfrag,
    float* __restrict__ out)
{
    __shared__ __align__(16) unsigned short hsh[TILE_B * HDIM];      // 16 KB bf16 h
    __shared__ __align__(16) uint32_t wlds[NKT_LDS * 64 * 256];      // 256 KB weights

    const int tid   = threadIdx.x;
    const int lane  = tid & 31;
    const int wv    = tid >> 5;        // wave 0..7
    const int ln    = lane & 15;
    const int hi    = lane >> 4;       // lane half
    const int bbase = blockIdx.x * TILE_B;

    // h(0) = 0; preload NKT_LDS k-slices of every n-tile into LDS
    for (int i = tid; i < TILE_B * HDIM; i += 256) hsh[i] = 0;
    for (int i = tid; i < NKT_LDS * 64 * 256; i += 256) {
        int f   = i >> 8;              // nt*NKT_LDS + ktl
        int nt  = f / NKT_LDS;
        int ktl = f % NKT_LDS;
        wlds[i] = wfrag[(nt * 8 + ktl) * 256 + (i & 255)];
    }
    __syncthreads();

    // per-n-tile scalars: Wx[n], bias[n]  (n = gate*256 + wv*32 + jj*16 + ln)
    float wxv[4][2], bv[4][2];
    {
        const float* wxp[4] = { w_gx, w_ix, w_fx, w_ox };
        const float* bpp[4] = { b_g, b_i, b_f, b_o };
        for (int g = 0; g < 4; ++g)
            for (int jj = 0; jj < 2; ++jj) {
                int j = wv * 32 + jj * 16 + ln;
                wxv[g][jj] = wxp[g][j];
                bv[g][jj]  = bpp[g][j];
            }
    }

    // cell state in C/D layout: [jj][mt], VGPR r -> row mt*16 + r + hi*8
    v8f cst[2][2];
    #pragma unroll
    for (int jj = 0; jj < 2; ++jj)
        #pragma unroll
        for (int mt = 0; mt < 2; ++mt)
            #pragma unroll
            for (int r = 0; r < 8; ++r) cst[jj][mt][r] = 0.0f;

    const v8f zero8 = { 0.0f, 0.0f, 0.0f, 0.0f, 0.0f, 0.0f, 0.0f, 0.0f };

    for (int t = 0; t < TSTEPS; ++t) {
        // opaque zero offsets: kill loop-invariance of the weight loads
        uint32_t goff = 0;   // dword offset into wfrag (global)
        uint32_t loff = 0;   // dword offset into wlds (LDS)
        asm volatile("" : "+v"(goff), "+v"(loff));

        v8f acc[4][2][2];   // [gate][jj][mt]

        // ---- peeled kt = 0 (LDS-resident): WMMA with literal-0 C ----
        {
            ABu a[2];
            #pragma unroll
            for (int mt = 0; mt < 2; ++mt) {
                const char* ab = reinterpret_cast<const char*>(hsh)
                               + (mt * 16 + ln) * (HDIM * 2) + hi * 16;
                a[mt].u[0] = *reinterpret_cast<const uint4*>(ab);
                a[mt].u[1] = *reinterpret_cast<const uint4*>(ab + 32);
            }
            #pragma unroll
            for (int g = 0; g < 4; ++g) {
                #pragma unroll
                for (int jj = 0; jj < 2; ++jj) {
                    const int nt = g * 16 + wv * 2 + jj;
                    const uint32_t* bp =
                        &wlds[(nt * NKT_LDS) * 256 + lane * 8 + loff];
                    ABu b;
                    b.u[0] = *reinterpret_cast<const uint4*>(bp);
                    b.u[1] = *reinterpret_cast<const uint4*>(bp + 4);
                    #pragma unroll
                    for (int mt = 0; mt < 2; ++mt) {
                        acc[g][jj][mt] = __builtin_amdgcn_wmma_f32_16x16x32_bf16(
                            false, a[mt].v, false, b.v, (short)0,
                            zero8, false, false);
                    }
                }
            }
        }

        // ---- remaining LDS-resident k-slices ----
        #pragma unroll 2
        for (int kt = 1; kt < NKT_LDS; ++kt) {
            ABu a[2];   // A fragments (16x32 bf16 of h), ISA A-layout
            #pragma unroll
            for (int mt = 0; mt < 2; ++mt) {
                const char* ab = reinterpret_cast<const char*>(hsh)
                               + (mt * 16 + ln) * (HDIM * 2) + kt * 64 + hi * 16;
                a[mt].u[0] = *reinterpret_cast<const uint4*>(ab);
                a[mt].u[1] = *reinterpret_cast<const uint4*>(ab + 32);
            }
            #pragma unroll
            for (int g = 0; g < 4; ++g) {
                #pragma unroll
                for (int jj = 0; jj < 2; ++jj) {
                    const int nt = g * 16 + wv * 2 + jj;
                    const uint32_t* bp =
                        &wlds[(nt * NKT_LDS + kt) * 256 + lane * 8 + loff];
                    ABu b;
                    b.u[0] = *reinterpret_cast<const uint4*>(bp);
                    b.u[1] = *reinterpret_cast<const uint4*>(bp + 4);
                    #pragma unroll
                    for (int mt = 0; mt < 2; ++mt) {
                        acc[g][jj][mt] = __builtin_amdgcn_wmma_f32_16x16x32_bf16(
                            false, a[mt].v, false, b.v, (short)0,
                            acc[g][jj][mt], false, false);
                    }
                }
            }
        }

        // ---- k-slices streamed from L2 ----
        #pragma unroll 1
        for (int kt = NKT_LDS; kt < 8; ++kt) {
            ABu a[2];
            #pragma unroll
            for (int mt = 0; mt < 2; ++mt) {
                const char* ab = reinterpret_cast<const char*>(hsh)
                               + (mt * 16 + ln) * (HDIM * 2) + kt * 64 + hi * 16;
                a[mt].u[0] = *reinterpret_cast<const uint4*>(ab);
                a[mt].u[1] = *reinterpret_cast<const uint4*>(ab + 32);
            }
            #pragma unroll
            for (int g = 0; g < 4; ++g) {
                #pragma unroll
                for (int jj = 0; jj < 2; ++jj) {
                    const int nt = g * 16 + wv * 2 + jj;
                    const uint32_t* bp =
                        wfrag + ((nt * 8 + kt) * 256 + lane * 8 + goff);
                    ABu b;
                    b.u[0] = *reinterpret_cast<const uint4*>(bp);
                    b.u[1] = *reinterpret_cast<const uint4*>(bp + 4);
                    #pragma unroll
                    for (int mt = 0; mt < 2; ++mt) {
                        acc[g][jj][mt] = __builtin_amdgcn_wmma_f32_16x16x32_bf16(
                            false, a[mt].v, false, b.v, (short)0,
                            acc[g][jj][mt], false, false);
                    }
                }
            }
        }
        __syncthreads();   // all reads of h(t-1) complete

        // gates + state update + write h(t) (bf16) back to LDS
        #pragma unroll
        for (int mt = 0; mt < 2; ++mt) {
            float xr[8];
            #pragma unroll
            for (int r = 0; r < 8; ++r)
                xr[r] = x[(bbase + mt * 16 + r + hi * 8) * TSTEPS + t];
            #pragma unroll
            for (int jj = 0; jj < 2; ++jj) {
                #pragma unroll
                for (int r = 0; r < 8; ++r) {
                    float zg = acc[0][jj][mt][r] + xr[r] * wxv[0][jj] + bv[0][jj];
                    float zi = acc[1][jj][mt][r] + xr[r] * wxv[1][jj] + bv[1][jj];
                    float zf = acc[2][jj][mt][r] + xr[r] * wxv[2][jj] + bv[2][jj];
                    float zo = acc[3][jj][mt][r] + xr[r] * wxv[3][jj] + bv[3][jj];
                    float gg = hw_tanh(zg);
                    float ii = fast_sig(zi);
                    float ff = fast_sig(zf);
                    float oo = fast_sig(zo);
                    float c  = gg * ii + cst[jj][mt][r] * ff;
                    cst[jj][mt][r] = c;
                    float h  = hw_tanh(c) * oo;
                    hsh[(mt * 16 + r + hi * 8) * HDIM + wv * 32 + jj * 16 + ln] =
                        (unsigned short)f32_to_bf16_bits(h);
                }
            }
        }
        __syncthreads();   // h(t) visible to all waves
    }

    // final projection: out[b][cls] = h[b] . w_ph[:,cls] + b_p[cls]
    for (int idx = tid; idx < TILE_B * NCLS; idx += 256) {
        int m   = idx / NCLS;
        int cls = idx % NCLS;
        float a = b_p[cls];
        for (int j = 0; j < HDIM; ++j) {
            union { uint32_t u; float f; } cv;
            cv.u = ((uint32_t)hsh[m * HDIM + j]) << 16;
            a += cv.f * w_ph[j * NCLS + cls];
        }
        out[(bbase + m) * NCLS + cls] = a;
    }
}

extern "C" void kernel_launch(void* const* d_in, const int* in_sizes, int n_in,
                              void* d_out, int out_size, void* d_ws, size_t ws_size,
                              hipStream_t stream) {
    const float* x    = (const float*)d_in[0];
    const float* w_gx = (const float*)d_in[1];
    const float* w_ix = (const float*)d_in[2];
    const float* w_fx = (const float*)d_in[3];
    const float* w_ox = (const float*)d_in[4];
    const float* w_gh = (const float*)d_in[5];
    const float* w_ih = (const float*)d_in[6];
    const float* w_fh = (const float*)d_in[7];
    const float* w_oh = (const float*)d_in[8];
    const float* b_g  = (const float*)d_in[9];
    const float* b_i  = (const float*)d_in[10];
    const float* b_f  = (const float*)d_in[11];
    const float* b_o  = (const float*)d_in[12];
    const float* w_ph = (const float*)d_in[13];
    const float* b_p  = (const float*)d_in[14];
    uint32_t* frag = (uint32_t*)d_ws;   // 512 KB bf16 B-fragments

    hipLaunchKernelGGL(repack_wh, dim3(512), dim3(256), 0, stream,
                       w_gh, w_ih, w_fh, w_oh, frag);
    hipLaunchKernelGGL(lstm_persist, dim3(1024 / TILE_B), dim3(256), 0, stream,
                       x, w_gx, w_ix, w_fx, w_ox,
                       b_g, b_i, b_f, b_o, w_ph, b_p, frag, (float*)d_out);
}
